// Attention_54855322304634
// MI455X (gfx1250) — matrix operand order repaired and checked by reading, not compiled, not run on hardware
//
#include <hip/hip_runtime.h>

typedef __attribute__((ext_vector_type(16))) __bf16 v16bf;
typedef __attribute__((ext_vector_type(8)))  float  v8f;

// ---------------------------------------------------------------- helpers
__device__ __forceinline__ __bf16 f2bf(float f) {
  union { float f; unsigned int u; } v; v.f = f;
  unsigned int r = (v.u + 0x7FFFu + ((v.u >> 16) & 1u)) >> 16;
  union { unsigned short s; __bf16 b; } o; o.s = (unsigned short)r;
  return o.b;
}

__device__ __forceinline__ unsigned int pack2bf(float a, float b) {
  union { __bf16 h[2]; unsigned int u; } o;
  o.h[0] = f2bf(a); o.h[1] = f2bf(b);
  return o.u;
}

// elems 0..7 from p0 (8 contiguous bf16), elems 8..15 from p1
__device__ __forceinline__ v16bf load_frag(const __bf16* p0, const __bf16* p1) {
  union { uint4 u[2]; v16bf v; } r;
  r.u[0] = *(const uint4*)p0;
  r.u[1] = *(const uint4*)p1;
  return r.v;
}

__device__ __forceinline__ v8f wmma_bf16(v16bf a, v16bf b, v8f c) {
  return __builtin_amdgcn_wmma_f32_16x16x32_bf16(
      false, a, false, b, (short)0, c, false, false);
}

// gfx1250 async DMA: memory -> LDS, 16B per lane, tracked by ASYNCcnt
__device__ __forceinline__ void async_b128(unsigned lds_off, const void* gptr) {
  asm volatile("global_load_async_to_lds_b128 %0, %1, off"
               :: "v"(lds_off), "v"(gptr) : "memory");
}
__device__ __forceinline__ void wait_async0() {
  asm volatile("s_wait_asynccnt 0x0" ::: "memory");
}
// low 32 bits of a generic LDS pointer == LDS byte offset (ISA aperture rule)
__device__ __forceinline__ unsigned lds_off(const void* p) {
  return (unsigned)(unsigned long long)p;
}

#define H_  1024
#define S_  2048
#define NH_ 16
#define HD_ 64
#define TH3_ 3072

// ---------------------------------------------------------------- K1: mod = silu(ada) @ mod_w^T + mod_b
__global__ __launch_bounds__(256) void mod_kernel(
    const float* __restrict__ ada, const float* __restrict__ mod_w,
    const float* __restrict__ mod_b, float* __restrict__ mod) {
  int idx = blockIdx.x * 256 + threadIdx.x;       // b*3H + j
  if (idx >= 2 * TH3_) return;
  int b = idx / TH3_, j = idx % TH3_;
  const float* a = ada + (size_t)b * H_;
  const float* w = mod_w + (size_t)j * H_;
  float acc = 0.f;
  for (int t = 0; t < H_; t += 4) {
    float4 av = *(const float4*)(a + t);
    float4 wv = *(const float4*)(w + t);
    acc += (av.x / (1.f + __expf(-av.x))) * wv.x;
    acc += (av.y / (1.f + __expf(-av.y))) * wv.y;
    acc += (av.z / (1.f + __expf(-av.z))) * wv.z;
    acc += (av.w / (1.f + __expf(-av.w))) * wv.w;
  }
  mod[idx] = acc + mod_b[j];
}

// ---------------------------------------------------------------- K2: f32 -> bf16 convert (weights)
__global__ __launch_bounds__(256) void cvt_bf16_kernel(
    const float* __restrict__ in, __bf16* __restrict__ out, int n4) {
  int i = blockIdx.x * 256 + threadIdx.x;
  if (i >= n4) return;
  float4 v = *(const float4*)(in + (size_t)i * 4);
  union { __bf16 h[4]; uint2 u; } o;
  o.h[0] = f2bf(v.x); o.h[1] = f2bf(v.y); o.h[2] = f2bf(v.z); o.h[3] = f2bf(v.w);
  *(uint2*)(out + (size_t)i * 4) = o.u;
}

// ---------------------------------------------------------------- K3: LN(x) * (scale+1) + shift -> bf16
__global__ __launch_bounds__(256) void ln_mod_kernel(
    const float* __restrict__ x, const float* __restrict__ ln_w,
    const float* __restrict__ mod, __bf16* __restrict__ h) {
  int row = blockIdx.x;                 // b*S + s
  int b = row >> 11;
  const float* xr = x + (size_t)row * H_;
  int t = threadIdx.x;
  float4 v = *(const float4*)(xr + t * 4);
  float s  = v.x + v.y + v.z + v.w;
  float sq = v.x*v.x + v.y*v.y + v.z*v.z + v.w*v.w;
  for (int m = 16; m >= 1; m >>= 1) { s += __shfl_xor(s, m, 32); sq += __shfl_xor(sq, m, 32); }
  __shared__ float ws[8], wq[8];
  int wave = t >> 5, lane = t & 31;
  if (lane == 0) { ws[wave] = s; wq[wave] = sq; }
  __syncthreads();
  if (t == 0) { float a = 0, c = 0; for (int i = 0; i < 8; ++i) { a += ws[i]; c += wq[i]; } ws[0] = a; wq[0] = c; }
  __syncthreads();
  float mean = ws[0] * (1.f / 1024.f);
  float var  = wq[0] * (1.f / 1024.f) - mean * mean;
  float inv  = rsqrtf(var + 1e-5f);
  const float* mb = mod + (size_t)b * TH3_;
  float4 lw = *(const float4*)(ln_w + t * 4);
  float4 sc = *(const float4*)(mb + t * 4);
  float4 sh = *(const float4*)(mb + H_ + t * 4);
  union { __bf16 hv[4]; uint2 u; } o;
  o.hv[0] = f2bf((v.x - mean) * inv * lw.x * (sc.x + 1.f) + sh.x);
  o.hv[1] = f2bf((v.y - mean) * inv * lw.y * (sc.y + 1.f) + sh.y);
  o.hv[2] = f2bf((v.z - mean) * inv * lw.z * (sc.z + 1.f) + sh.z);
  o.hv[3] = f2bf((v.w - mean) * inv * lw.w * (sc.w + 1.f) + sh.w);
  *(uint2*)(h + (size_t)row * H_ + t * 4) = o.u;
}

// ---------------------------------------------------------------- K4/K7: tiled WMMA GEMM  C = A(MxK) * B(NxK)^T
// block tile 128x128, 8 waves (2 x 4), each wave 64x32 (4x2 subtiles), K-step 32
// tiles staged with gfx1250 async memory->LDS DMA (ASYNCcnt)
__global__ __launch_bounds__(256) void gemm_bf16_kernel(
    const __bf16* __restrict__ A, const __bf16* __restrict__ B,
    float* __restrict__ C, int M, int N, int K,
    const float* __restrict__ gate, int S) {
  __shared__ alignas(16) __bf16 At[128 * 32];
  __shared__ alignas(16) __bf16 Bt[128 * 32];
  const int tid = threadIdx.x, wave = tid >> 5, lane = tid & 31;
  const int wm = wave >> 2, wn = wave & 3;
  const int m = lane & 15, hi = lane >> 4;
  const int row0 = blockIdx.y * 128, col0 = blockIdx.x * 128;

  // per-thread staging addresses: chunk tid covers (row=tid>>2, col8=(tid&3)*8),
  // second chunk is 64 rows below; LDS offsets are linear 16B chunks.
  const int ar = tid >> 2, ac = (tid & 3) * 8;
  const __bf16* agp0 = A + (size_t)(row0 + ar) * K + ac;
  const __bf16* agp1 = A + (size_t)(row0 + ar + 64) * K + ac;
  const __bf16* bgp0 = B + (size_t)(col0 + ar) * K + ac;
  const __bf16* bgp1 = B + (size_t)(col0 + ar + 64) * K + ac;
  const unsigned lA = lds_off(&At[0]) + tid * 16;
  const unsigned lB = lds_off(&Bt[0]) + tid * 16;

  v8f acc[4][2] = {};
  for (int k0 = 0; k0 < K; k0 += 32) {
    __syncthreads();                       // previous tile fully consumed
    async_b128(lA,        agp0 + k0);
    async_b128(lA + 4096, agp1 + k0);
    async_b128(lB,        bgp0 + k0);
    async_b128(lB + 4096, bgp1 + k0);
    wait_async0();
    __syncthreads();                       // tiles visible to all waves

    v16bf af[4], bfr[2];
    for (int ms = 0; ms < 4; ++ms) {
      const __bf16* p = &At[(wm * 64 + ms * 16 + m) * 32];
      af[ms] = load_frag(p + hi * 16, p + hi * 16 + 8);     // A: contiguous k = 16*hi + s (same as B)
    }
    for (int ns = 0; ns < 2; ++ns) {
      const __bf16* p = &Bt[(wn * 32 + ns * 16 + m) * 32 + hi * 16];
      bfr[ns] = load_frag(p, p + 8);                        // B: contiguous k = 16*hi + s (same as A)
    }
    for (int ms = 0; ms < 4; ++ms)
      for (int ns = 0; ns < 2; ++ns)
        acc[ms][ns] = wmma_bf16(af[ms], bfr[ns], acc[ms][ns]);
  }
  const bool useg = (gate != nullptr);
  for (int ms = 0; ms < 4; ++ms)
    for (int ns = 0; ns < 2; ++ns)
      for (int r = 0; r < 8; ++r) {
        int row = row0 + wm * 64 + ms * 16 + hi * 8 + r;
        int col = col0 + wn * 32 + ns * 16 + m;
        float v = acc[ms][ns][r];
        if (useg) v *= gate[(row / S) * TH3_ + 2 * H_ + col];
        C[(size_t)row * N + col] = v;
      }
}

// ---------------------------------------------------------------- K5: per-head QK layernorm + RoPE + bf16 split
__global__ __launch_bounds__(256) void qkv_post_kernel(
    const float* __restrict__ qkv, const float* __restrict__ freqs,
    const float* __restrict__ qn_w, const float* __restrict__ kn_w,
    __bf16* __restrict__ qo, __bf16* __restrict__ ko, __bf16* __restrict__ vo) {
  int gw = blockIdx.x * 8 + (threadIdx.x >> 5);   // (b, head, s) row
  int lane = threadIdx.x & 31;
  int s = gw & (S_ - 1), head = (gw >> 11) & 15, b = gw >> 15;
  const float* qp = qkv + ((size_t)(b * S_ + s)) * TH3_ + head * HD_;
  float2 qv = *(const float2*)(qp + 2 * lane);
  float2 kv = *(const float2*)(qp + H_ + 2 * lane);
  float2 vv = *(const float2*)(qp + 2 * H_ + 2 * lane);

  float sq_s = qv.x + qv.y,           kq_s = kv.x + kv.y;
  float sq_q = qv.x*qv.x + qv.y*qv.y, kq_q = kv.x*kv.x + kv.y*kv.y;
  for (int msk = 16; msk >= 1; msk >>= 1) {
    sq_s += __shfl_xor(sq_s, msk, 32); sq_q += __shfl_xor(sq_q, msk, 32);
    kq_s += __shfl_xor(kq_s, msk, 32); kq_q += __shfl_xor(kq_q, msk, 32);
  }
  float qm = sq_s * (1.f / 64.f), km = kq_s * (1.f / 64.f);
  float qi = rsqrtf(sq_q * (1.f / 64.f) - qm * qm + 1e-5f);
  float ki = rsqrtf(kq_q * (1.f / 64.f) - km * km + 1e-5f);
  float2 w0 = *(const float2*)(qn_w + 2 * lane);
  float2 w1 = *(const float2*)(kn_w + 2 * lane);
  float q0 = (qv.x - qm) * qi * w0.x, q1 = (qv.y - qm) * qi * w0.y;
  float k0 = (kv.x - km) * ki * w1.x, k1 = (kv.y - km) * ki * w1.y;
  float2 f = *(const float2*)(freqs + ((size_t)s * 32 + lane) * 2);
  float qre = q0 * f.x - q1 * f.y, qim = q1 * f.x + q0 * f.y;
  float kre = k0 * f.x - k1 * f.y, kim = k1 * f.x + k0 * f.y;
  size_t oi = (((size_t)(b * NH_ + head)) * S_ + s) * HD_ + 2 * lane;
  *(unsigned int*)(qo + oi) = pack2bf(qre, qim);
  *(unsigned int*)(ko + oi) = pack2bf(kre, kim);
  *(unsigned int*)(vo + oi) = pack2bf(vv.x, vv.y);
}

// ---------------------------------------------------------------- K6: flash attention (non-causal), WMMA bf16
// grid (S/128, B*nh); 8 waves, wave owns 16 query rows; 64-key tiles
__global__ __launch_bounds__(256) void flash_kernel(
    const __bf16* __restrict__ q, const __bf16* __restrict__ k,
    const __bf16* __restrict__ v, __bf16* __restrict__ o) {
  __shared__ alignas(16) __bf16 Kt[64 * 64];        // [key][d]
  __shared__ alignas(16) __bf16 Vt[64 * 64];        // [d-col][key]  (transposed)
  __shared__ alignas(16) __bf16 Pb[8 * 16 * 64];    // per-wave P tile [q16][key64]
  const int bh = blockIdx.y;
  const int tid = threadIdx.x, wave = tid >> 5, lane = tid & 31;
  const int m = lane & 15, hi = lane >> 4;
  const size_t hoff = (size_t)bh * S_ * HD_;
  const __bf16* qh = q + hoff;
  const __bf16* kh = k + hoff;
  const __bf16* vh = v + hoff;
  const int qbase = blockIdx.x * 128 + wave * 16;

  v16bf aq[2];
  for (int c = 0; c < 2; ++c) {
    const __bf16* p = qh + (size_t)(qbase + m) * HD_ + c * 32 + hi * 16;
    aq[c] = load_frag(p, p + 8);
  }
  v8f acc[4] = {};
  float mrow[8], lrow[8];
  for (int r = 0; r < 8; ++r) { mrow[r] = -1e30f; lrow[r] = 0.f; }

  // staging addresses: chunk tid -> (row tid>>3, col8 (tid&7)*8); chunk2 = +32 rows
  const int kr = tid >> 3, kc = (tid & 7) * 8;
  const __bf16* kgp0 = kh + (size_t)kr * HD_ + kc;
  const __bf16* vgp0 = vh + (size_t)kr * HD_ + kc;
  const unsigned lK = lds_off(&Kt[0]) + tid * 16;

  for (int j0 = 0; j0 < S_; j0 += 64) {
    __syncthreads();
    // K tile via async DMA (row major)
    async_b128(lK,        kgp0 + (size_t)j0 * HD_);
    async_b128(lK + 4096, kgp0 + (size_t)(j0 + 32) * HD_);
    // V tile: manual transpose into LDS
    {
      uint4 d0 = *(const uint4*)(vgp0 + (size_t)j0 * HD_);
      uint4 d1 = *(const uint4*)(vgp0 + (size_t)(j0 + 32) * HD_);
      const __bf16* p0 = (const __bf16*)&d0;
      const __bf16* p1 = (const __bf16*)&d1;
      for (int e = 0; e < 8; ++e) Vt[(kc + e) * 64 + kr] = p0[e];
      for (int e = 0; e < 8; ++e) Vt[(kc + e) * 64 + kr + 32] = p1[e];
    }
    wait_async0();
    __syncthreads();

    // scores S = Q K^T  (16 x 64)
    v8f sacc[4];
    for (int nt = 0; nt < 4; ++nt) {
      const __bf16* p = &Kt[(nt * 16 + m) * 64 + hi * 16];
      v16bf b0 = load_frag(p, p + 8);
      v16bf b1 = load_frag(p + 32, p + 40);
      v8f cz = {};
      cz = wmma_bf16(aq[0], b0, cz);
      sacc[nt] = wmma_bf16(aq[1], b1, cz);
    }
    // online softmax (rows live across the 16-lane half, elem r)
    for (int r = 0; r < 8; ++r) {
      float vmax = fmaxf(fmaxf(sacc[0][r], sacc[1][r]), fmaxf(sacc[2][r], sacc[3][r])) * 0.125f;
      for (int msk = 8; msk >= 1; msk >>= 1) vmax = fmaxf(vmax, __shfl_xor(vmax, msk, 32));
      float mn = fmaxf(mrow[r], vmax);
      float corr = __expf(mrow[r] - mn);
      float rsum = 0.f;
      for (int nt = 0; nt < 4; ++nt) {
        float p = __expf(sacc[nt][r] * 0.125f - mn);
        sacc[nt][r] = p;
        rsum += p;
      }
      for (int msk = 8; msk >= 1; msk >>= 1) rsum += __shfl_xor(rsum, msk, 32);
      lrow[r] = lrow[r] * corr + rsum;
      mrow[r] = mn;
      for (int nt = 0; nt < 4; ++nt) acc[nt][r] *= corr;
    }
    // P (C layout, f32) -> LDS bf16 -> A-layout frags
    __bf16* pw = &Pb[wave * 16 * 64];
    for (int nt = 0; nt < 4; ++nt)
      for (int r = 0; r < 8; ++r)
        pw[(hi * 8 + r) * 64 + nt * 16 + m] = f2bf(sacc[nt][r]);
    v16bf pa[2];
    for (int c = 0; c < 2; ++c) {
      const __bf16* p = &pw[m * 64 + c * 32 + hi * 16];
      pa[c] = load_frag(p, p + 8);
    }
    // O += P V   (V fetched from transposed tile: contiguous along key)
    for (int nt = 0; nt < 4; ++nt) {
      const __bf16* p = &Vt[(nt * 16 + m) * 64 + hi * 16];
      v16bf b0 = load_frag(p, p + 8);
      v16bf b1 = load_frag(p + 32, p + 40);
      acc[nt] = wmma_bf16(pa[0], b0, acc[nt]);
      acc[nt] = wmma_bf16(pa[1], b1, acc[nt]);
    }
  }

  // normalize + scatter to o[B,S,H] bf16
  const int b = bh >> 4, head = bh & 15;
  for (int nt = 0; nt < 4; ++nt)
    for (int r = 0; r < 8; ++r) {
      float ov = acc[nt][r] / lrow[r];
      int srow = qbase + hi * 8 + r;
      int col = head * HD_ + nt * 16 + m;
      o[((size_t)b * S_ + srow) * H_ + col] = f2bf(ov);
    }
}

// ---------------------------------------------------------------- launcher
extern "C" void kernel_launch(void* const* d_in, const int* in_sizes, int n_in,
                              void* d_out, int out_size, void* d_ws, size_t ws_size,
                              hipStream_t stream) {
  const float* x      = (const float*)d_in[0];
  const float* ada    = (const float*)d_in[1];
  const float* freqs  = (const float*)d_in[2];
  const float* w_qkv  = (const float*)d_in[3];
  const float* w_o    = (const float*)d_in[4];
  const float* ln_w   = (const float*)d_in[5];
  const float* mod_w  = (const float*)d_in[6];
  const float* mod_b  = (const float*)d_in[7];
  const float* qn_w   = (const float*)d_in[8];
  const float* kn_w   = (const float*)d_in[9];
  float* out = (float*)d_out;

  char* ws = (char*)d_ws;
  auto take = [&](size_t bytes) { char* p = ws; ws += (bytes + 255) & ~size_t(255); return p; };
  float*  mod      = (float*) take((size_t)2 * TH3_ * 4);
  __bf16* w_qkv_bf = (__bf16*)take((size_t)TH3_ * H_ * 2);
  __bf16* w_o_bf   = (__bf16*)take((size_t)H_ * H_ * 2);
  __bf16* h_bf     = (__bf16*)take((size_t)2 * S_ * H_ * 2);
  float*  qkv      = (float*) take((size_t)2 * S_ * TH3_ * 4);
  __bf16* q_bf     = (__bf16*)take((size_t)2 * S_ * H_ * 2);
  __bf16* k_bf     = (__bf16*)take((size_t)2 * S_ * H_ * 2);
  __bf16* v_bf     = (__bf16*)take((size_t)2 * S_ * H_ * 2);
  __bf16* o_bf     = (__bf16*)take((size_t)2 * S_ * H_ * 2);

  // 1. AdaLN modulation vector
  mod_kernel<<<(2 * TH3_ + 255) / 256, 256, 0, stream>>>(ada, mod_w, mod_b, mod);
  // 2. weight conversion to bf16
  cvt_bf16_kernel<<<(TH3_ * H_ / 4 + 255) / 256, 256, 0, stream>>>(w_qkv, w_qkv_bf, TH3_ * H_ / 4);
  cvt_bf16_kernel<<<(H_ * H_ / 4 + 255) / 256, 256, 0, stream>>>(w_o, w_o_bf, H_ * H_ / 4);
  // 3. LN + scale/shift -> h (bf16)
  ln_mod_kernel<<<2 * S_, 256, 0, stream>>>(x, ln_w, mod, h_bf);
  // 4. QKV GEMM: [4096,1024] x [3072,1024]^T -> f32
  gemm_bf16_kernel<<<dim3(TH3_ / 128, (2 * S_) / 128), 256, 0, stream>>>(
      h_bf, w_qkv_bf, qkv, 2 * S_, TH3_, H_, nullptr, S_);
  // 5. per-head QK LN + RoPE, split to q/k/v bf16
  qkv_post_kernel<<<(2 * NH_ * S_) / 8, 256, 0, stream>>>(
      qkv, freqs, qn_w, kn_w, q_bf, k_bf, v_bf);
  // 6. flash attention
  flash_kernel<<<dim3(S_ / 128, 2 * NH_), 256, 0, stream>>>(q_bf, k_bf, v_bf, o_bf);
  // 7. output projection * gate -> f32 d_out
  gemm_bf16_kernel<<<dim3(H_ / 128, (2 * S_) / 128), 256, 0, stream>>>(
      o_bf, w_o_bf, out, 2 * S_, H_, H_, mod, S_);
}
